// SAShiftedWindowAttention_69758858822051
// MI455X (gfx1250) — matrix operand order repaired and checked by reading, not compiled, not run on hardware
//
#include <hip/hip_runtime.h>
#include <hip/hip_bf16.h>

// ---------------------------------------------------------------------------
// Fused Swin shifted-window attention for MI455X (gfx1250, wave32, WMMA).
//
// Pre-pass: convert qkv/proj weights fp32 -> f16 ONCE into d_ws (shared by all
// 8192 window-blocks through the 192 MB L2).
// Main kernel: 1 block = 1 window (256 thr = 8 wave32), fully fused in LDS:
//   gather(+roll) -> QKV GEMM -> QK^T + rpb + mask -> softmax -> AV GEMM
//   -> proj GEMM -> scatter(+unroll), all matmuls v_wmma_f32_16x16x32_f16.
// All LDS strides are 16B multiples so fragments move as b128 vectors.
// Phase-1 tile mapping nt = ni*8 + wave makes the q/k/v epilogue selection a
// compile-time decision per unrolled iteration (no exec-mask branching).
// ---------------------------------------------------------------------------

typedef __attribute__((ext_vector_type(16))) _Float16 v16h;
typedef __attribute__((ext_vector_type(8)))  _Float16 v8h;
typedef __attribute__((ext_vector_type(4)))  _Float16 v4h;
typedef __attribute__((ext_vector_type(4)))  float    v4f;
typedef __attribute__((ext_vector_type(8)))  float    v8f;

#define WSZ   7
#define SHIFT 3
#define NHEAD 4
#define CDIM  128
#define HDIM  32
#define NTOK  49
#define NPAD  64
#define IMGH  112
#define IMGW  112
#define NWIN  16

#define XSTR  136      // halves; 272 B rows (16B multiple)
#define QKSTR 40       // halves; 80 B rows  (16B multiple)
#define VTSTR 72       // halves; 144 B rows (16B multiple)
#define ASTR  68       // floats; 272 B rows
#define A16S  64       // halves; 128 B rows

// ---- LDS layout (all offsets 16B aligned; phase-ordered aliasing) ----------
#define SZ_X    (NPAD * XSTR * 2)                 // 17408  (sO aliases later)
#define OFF_QK  (SZ_X)
#define SZ_QK   (2 * NHEAD * NPAD * QKSTR * 2)    // 40960  (q,k row-major)
#define OFF_VT  (OFF_QK + SZ_QK)                  // 58368
#define SZ_VT   (NHEAD * HDIM * VTSTR * 2)        // 18432  (v transposed)
#define OFF_ATT (OFF_VT + SZ_VT)                  // 76800
#define SZ_ATT  (NHEAD * NPAD * ASTR * 4)         // 69632  (f16 probs alias)
#define OFF_RPB (OFF_ATT + SZ_ATT)                // 146432
#define SZ_RPB  (169 * NHEAD * 4)                 // 2704
#define SMEMB   (OFF_RPB + SZ_RPB)                // 149136 B

#define NQKV (3 * CDIM * CDIM)                    // 49152 qkv weight elems
#define NPRJ (CDIM * CDIM)                        // 16384 proj weight elems

// ---- WMMA fragment loaders per CDNA5 ISA 7.12.2 (wave32) -------------------

// A fragment 16x32 f16 from row-major [m][k]: per lane one contiguous 32B run.
__device__ __forceinline__ v16h frag_a(const _Float16* base, int stride, int lane) {
  const int m = lane & 15;
  const int g = (lane >> 4) & 1;
  const _Float16* r = base + m * stride + g * 16;
  v8h lo = *(const v8h*)(r);
  v8h hi = *(const v8h*)(r + 8);
  v16h a;
#pragma unroll
  for (int i = 0; i < 8; ++i) { a[i] = lo[i]; a[8 + i] = hi[i]; }
  return a;
}

// B fragment from B^T row-major (bt[n][k], f16 in LDS): one 32B run per lane.
__device__ __forceinline__ v16h frag_bt(const _Float16* base, int stride, int lane) {
  const _Float16* r = base + (lane & 15) * stride + ((lane >> 4) & 1) * 16;
  v8h lo = *(const v8h*)(r);
  v8h hi = *(const v8h*)(r + 8);
  v16h b;
#pragma unroll
  for (int i = 0; i < 8; ++i) { b[i] = lo[i]; b[8 + i] = hi[i]; }
  return b;
}

// B fragment from pre-converted f16 weight row in global (W[n][k] == B^T).
__device__ __forceinline__ v16h frag_bw(const _Float16* rowbase, int lane) {
  const _Float16* r = rowbase + ((lane >> 4) & 1) * 16;
  v8h lo = *(const v8h*)(r);
  v8h hi = *(const v8h*)(r + 8);
  v16h b;
#pragma unroll
  for (int i = 0; i < 8; ++i) { b[i] = lo[i]; b[8 + i] = hi[i]; }
  return b;
}

__device__ __forceinline__ int region(int idx) {
  return (idx >= IMGH - SHIFT) ? 2 : ((idx >= IMGH - WSZ) ? 1 : 0);
}

// ---------------------------------------------------------------------------
// Pre-pass: fp32 -> f16 weight conversion (runs once per launch, 256 blocks).
// ---------------------------------------------------------------------------
extern "C" __global__ void __launch_bounds__(256)
swin_wcvt_kernel(const float* __restrict__ qkvw, const float* __restrict__ projw,
                 _Float16* __restrict__ w16) {
  int i = blockIdx.x * 256 + threadIdx.x;
  if (i < NQKV)              w16[i] = (_Float16)qkvw[i];
  else if (i < NQKV + NPRJ)  w16[i] = (_Float16)projw[i - NQKV];
}

// ---------------------------------------------------------------------------
// Main fused kernel.
// ---------------------------------------------------------------------------
extern "C" __global__ void __launch_bounds__(256)
swin_window_attn_kernel(const float* __restrict__ x,
                        const _Float16* __restrict__ w16,   // [qkv f16 | proj f16]
                        const float* __restrict__ qkvb,
                        const float* __restrict__ projb,
                        const float* __restrict__ rpbt,
                        float* __restrict__ out)
{
  __shared__ __align__(16) unsigned char smem[SMEMB];
  _Float16* sX   = (_Float16*)(smem);            // phase 0-1
  _Float16* sQK  = (_Float16*)(smem + OFF_QK);   // phase 1-3: q,k [tok][d]
  _Float16* sVT  = (_Float16*)(smem + OFF_VT);   // phase 1-3: v  [d][tok]
  float*    sAtt = (float*)   (smem + OFF_ATT);  // phase 2: f32 scores
  _Float16* sP16 = (_Float16*)(smem + OFF_ATT);  // phase 2b-3: f16 probs (alias)
  _Float16* sO   = (_Float16*)(smem);            // phase 3-4 (aliases sX)
  float*    sRPB = (float*)   (smem + OFF_RPB);

  const _Float16* qkvw16 = w16;
  const _Float16* prjw16 = w16 + NQKV;

  const int tid  = threadIdx.x;
  const int lane = tid & 31;
  const int wave = tid >> 5;
  const int g    = (lane >> 4) & 1;

  const int blk = blockIdx.x;
  const int win = blk & (NWIN * NWIN - 1);
  const int bb  = blk >> 8;
  const int wy  = win >> 4;
  const int wx  = win & 15;
  const bool edge = (wy == NWIN - 1) || (wx == NWIN - 1);  // uniform: mask != 0

  __builtin_prefetch(qkvw16 + (tid << 5), 0, 1);
  __builtin_prefetch(prjw16 + (tid << 4), 0, 1);

  // ---------------- phase 0: gather shifted window -> f16 LDS ---------------
#pragma unroll
  for (int e = 0; e < 8; ++e) {                 // 8 x float4 per thread
    int gidx = (tid + e * 256) * 4;             // covers 64*128 elements
    int row = gidx >> 7;
    int col = gidx & 127;
    v4f v = {0.f, 0.f, 0.f, 0.f};
    if (row < NTOK) {
      int wi = row / WSZ, wj = row % WSZ;
      int gr = (wy * WSZ + wi + SHIFT) % IMGH;  // xs = roll(x, -SHIFT)
      int gc = (wx * WSZ + wj + SHIFT) % IMGW;
      v = *(const v4f*)(x + (((size_t)bb * IMGH + gr) * IMGW + gc) * CDIM + col);
    }
    v4h pk;
    pk[0] = (_Float16)v[0]; pk[1] = (_Float16)v[1];
    pk[2] = (_Float16)v[2]; pk[3] = (_Float16)v[3];
    *(v4h*)(sX + row * XSTR + col) = pk;
  }
  for (int i = tid; i < 169 * NHEAD; i += 256) sRPB[i] = rpbt[i];
  __syncthreads();

  // ---------------- phase 1: QKV GEMM (64x128 @ 128x384) --------------------
  // ni = 0 -> q tiles, ni = 1 -> k tiles, ni = 2 -> v tiles (compile-time).
  {
    const float qscale = 0.17677669529663687f;   // 32^-0.5
    const int hh = wave >> 1;                    // head of this wave's tile
    const int dd = (wave & 1) * 16 + (lane & 15);// dim within head
#pragma unroll
    for (int ni = 0; ni < 3; ++ni) {
      const int nt = ni * 8 + wave;              // 24 N-tiles / 8 waves
      const int nG = nt * 16 + (lane & 15);
      const _Float16* wrow = qkvw16 + (size_t)nG * CDIM;
      v16h b0 = frag_bw(wrow +  0, lane);        // hoisted across 4 M-tiles
      v16h b1 = frag_bw(wrow + 32, lane);
      v16h b2 = frag_bw(wrow + 64, lane);
      v16h b3 = frag_bw(wrow + 96, lane);
      const float bias = qkvb[nG];
#pragma unroll
      for (int mt = 0; mt < 4; ++mt) {
        v8f acc = {};
        acc = __builtin_amdgcn_wmma_f32_16x16x32_f16(false, frag_a(sX + mt*16*XSTR +  0, XSTR, lane), false, b0, (short)0, acc, false, false);
        acc = __builtin_amdgcn_wmma_f32_16x16x32_f16(false, frag_a(sX + mt*16*XSTR + 32, XSTR, lane), false, b1, (short)0, acc, false, false);
        acc = __builtin_amdgcn_wmma_f32_16x16x32_f16(false, frag_a(sX + mt*16*XSTR + 64, XSTR, lane), false, b2, (short)0, acc, false, false);
        acc = __builtin_amdgcn_wmma_f32_16x16x32_f16(false, frag_a(sX + mt*16*XSTR + 96, XSTR, lane), false, b3, (short)0, acc, false, false);
        if (ni < 2) {                            // q/k: row-major [token][d]
          const float sc = (ni == 0) ? qscale : 1.0f;   // compile-time select
          _Float16* dst = sQK + ((ni * NHEAD + hh) * NPAD) * QKSTR + dd;
#pragma unroll
          for (int r = 0; r < 8; ++r)
            dst[(mt * 16 + g * 8 + r) * QKSTR] = (_Float16)((acc[r] + bias) * sc);
        } else {                                 // v: transposed [d][token]
          v8h pk;
#pragma unroll
          for (int r = 0; r < 8; ++r) pk[r] = (_Float16)(acc[r] + bias);
          *(v8h*)(sVT + (hh * HDIM + dd) * VTSTR + mt * 16 + g * 8) = pk;
        }
      }
    }
  }
  __syncthreads();

  // ---------------- phase 2: scores = q @ k^T + rpb (+ mask on edge) --------
  {
    const int h = wave >> 1, sub = wave & 1;     // 2 waves per head
    const _Float16* qh = sQK + ((0 * NHEAD + h) * NPAD) * QKSTR;
    const _Float16* kh = sQK + ((1 * NHEAD + h) * NPAD) * QKSTR;
    const int oy = wy * WSZ, ox = wx * WSZ;
#pragma unroll
    for (int mi = 0; mi < 2; ++mi) {
      const int mt = sub * 2 + mi;
      v16h a = frag_a(qh + mt * 16 * QKSTR, QKSTR, lane);   // hoisted over nt
      const int pbase = mt * 16 + g * 8;
      int  pcode[8]; int pcnt[8]; bool pval[8];
#pragma unroll
      for (int r = 0; r < 8; ++r) {
        const int p = pbase + r;
        const int pi = p / WSZ, pj = p % WSZ;
        pcode[r] = pi * 13 + pj;
        pval[r]  = (p < NTOK);
        pcnt[r]  = edge ? (region(oy + pi) * 3 + region(ox + pj)) : 0;
      }
#pragma unroll
      for (int nt = 0; nt < 4; ++nt) {
        v16h bf = frag_bt(kh + nt * 16 * QKSTR, QKSTR, lane);
        v8f acc = {};
        acc = __builtin_amdgcn_wmma_f32_16x16x32_f16(false, a, false, bf, (short)0, acc, false, false);
        const int qtok = nt * 16 + (lane & 15);
        const bool qval = (qtok < NTOK);
        const int qi = qtok / WSZ, qj = qtok % WSZ;
        const int qcode = qi * 13 + qj;
        const int cq = edge ? (region(oy + qi) * 3 + region(ox + qj)) : 0;
#pragma unroll
        for (int r = 0; r < 8; ++r) {
          float v = acc[r];
          if (pval[r] && qval) {
            v += sRPB[(pcode[r] - qcode + 84) * NHEAD + h];
            if (edge && (pcnt[r] != cq)) v -= 100.0f;
          }
          sAtt[(h * NPAD + pbase + r) * ASTR + qtok] = v;
        }
      }
    }
  }
  __syncthreads();

  // ---------------- phase 2b: row softmax, f32 scores -> f16 probs ----------
  {
    const int h = tid >> 6;
    const int p = tid & 63;
    const float* rowf = sAtt + (h * NPAD + p) * ASTR;
    float rv[NTOK];
#pragma unroll
    for (int qb = 0; qb < 48; qb += 4) {         // vector reads (16B aligned)
      v4f t = *(const v4f*)(rowf + qb);
      rv[qb] = t[0]; rv[qb+1] = t[1]; rv[qb+2] = t[2]; rv[qb+3] = t[3];
    }
    rv[48] = rowf[48];
    float m = -1e30f;
#pragma unroll
    for (int q = 0; q < NTOK; ++q) m = fmaxf(m, rv[q]);
    float s = 0.f;
#pragma unroll
    for (int q = 0; q < NTOK; ++q) { rv[q] = __expf(rv[q] - m); s += rv[q]; }
    const float inv = 1.0f / s;
    __syncthreads();                 // all f32 rows consumed into registers
    _Float16* row16 = sP16 + (h * NPAD + p) * A16S;
#pragma unroll
    for (int qb = 0; qb < NPAD; qb += 8) {
      v8h pk;
#pragma unroll
      for (int j = 0; j < 8; ++j) {
        const int q = qb + j;
        pk[j] = (q < NTOK) ? (_Float16)(rv[q] * inv) : (_Float16)0.f;
      }
      *(v8h*)(row16 + qb) = pk;                  // pad cols zeroed
    }
  }
  __syncthreads();

  // ---------------- phase 3: out = probs @ v  (64x64 @ 64x32 per head) ------
  {
    const int h = wave >> 1, sub = wave & 1;
    const _Float16* ah = sP16 + (h * NPAD) * A16S;
    const _Float16* vt = sVT + (h * HDIM + sub * 16) * VTSTR;
    v16h b0 = frag_bt(vt +  0, VTSTR, lane);     // hoisted over 4 M-tiles
    v16h b1 = frag_bt(vt + 32, VTSTR, lane);
#pragma unroll
    for (int mt = 0; mt < 4; ++mt) {
      v8f acc = {};
      acc = __builtin_amdgcn_wmma_f32_16x16x32_f16(false, frag_a(ah + mt*16*A16S +  0, A16S, lane), false, b0, (short)0, acc, false, false);
      acc = __builtin_amdgcn_wmma_f32_16x16x32_f16(false, frag_a(ah + mt*16*A16S + 32, A16S, lane), false, b1, (short)0, acc, false, false);
      const int colG = h * HDIM + sub * 16 + (lane & 15);
#pragma unroll
      for (int r = 0; r < 8; ++r)
        sO[(mt * 16 + g * 8 + r) * XSTR + colG] = (_Float16)acc[r];
    }
  }
  __syncthreads();

  // ---------------- phase 4: proj GEMM + bias, scatter with un-roll ---------
  {
    const int nt = wave;                          // 8 N-tiles / 8 waves
    const int nG = nt * 16 + (lane & 15);
    const _Float16* wrow = prjw16 + (size_t)nG * CDIM;
    v16h b0 = frag_bw(wrow +  0, lane);           // hoisted over 4 M-tiles
    v16h b1 = frag_bw(wrow + 32, lane);
    v16h b2 = frag_bw(wrow + 64, lane);
    v16h b3 = frag_bw(wrow + 96, lane);
    const float bias = projb[nG];
#pragma unroll
    for (int mt = 0; mt < 4; ++mt) {
      v8f acc = {};
      acc = __builtin_amdgcn_wmma_f32_16x16x32_f16(false, frag_a(sO + mt*16*XSTR +  0, XSTR, lane), false, b0, (short)0, acc, false, false);
      acc = __builtin_amdgcn_wmma_f32_16x16x32_f16(false, frag_a(sO + mt*16*XSTR + 32, XSTR, lane), false, b1, (short)0, acc, false, false);
      acc = __builtin_amdgcn_wmma_f32_16x16x32_f16(false, frag_a(sO + mt*16*XSTR + 64, XSTR, lane), false, b2, (short)0, acc, false, false);
      acc = __builtin_amdgcn_wmma_f32_16x16x32_f16(false, frag_a(sO + mt*16*XSTR + 96, XSTR, lane), false, b3, (short)0, acc, false, false);
#pragma unroll
      for (int r = 0; r < 8; ++r) {
        const int tok = mt * 16 + g * 8 + r;
        if (tok < NTOK) {
          const int ti = tok / WSZ, tj = tok % WSZ;
          const int gr = (wy * WSZ + ti + SHIFT) % IMGH;   // roll(+SHIFT)
          const int gc = (wx * WSZ + tj + SHIFT) % IMGW;
          out[(((size_t)bb * IMGH + gr) * IMGW + gc) * CDIM + nG] = acc[r] + bias;
        }
      }
    }
  }
}

extern "C" void kernel_launch(void* const* d_in, const int* in_sizes, int n_in,
                              void* d_out, int out_size, void* d_ws, size_t ws_size,
                              hipStream_t stream) {
  (void)in_sizes; (void)n_in; (void)out_size; (void)ws_size;
  const float* x     = (const float*)d_in[0];
  const float* qkvw  = (const float*)d_in[1];
  const float* qkvb  = (const float*)d_in[2];
  const float* projw = (const float*)d_in[3];
  const float* projb = (const float*)d_in[4];
  const float* rpbt  = (const float*)d_in[5];
  float* out = (float*)d_out;
  _Float16* w16 = (_Float16*)d_ws;               // needs 128 KB scratch

  // pre-pass: one-time fp32 -> f16 weight conversion into workspace
  swin_wcvt_kernel<<<dim3((NQKV + NPRJ) / 256), dim3(256), 0, stream>>>(qkvw, projw, w16);

  const int num_blocks = 32 * NWIN * NWIN;       // 8192 windows
  swin_window_attn_kernel<<<dim3(num_blocks), dim3(256), 0, stream>>>(
      x, w16, qkvb, projb, rpbt, out);
}
